// Encode_420906795123
// MI455X (gfx1250) — compile-verified, matrix operand order checked by reading, not run on hardware
//
#include <hip/hip_runtime.h>
#include <hip/hip_bf16.h>

typedef __attribute__((ext_vector_type(16))) __bf16 bf16x16;
typedef __attribute__((ext_vector_type(8)))  float  v8f;
typedef __attribute__((ext_vector_type(4)))  int    v4i;
typedef unsigned short u16;
typedef unsigned int   u32;

#define HID    512
#define GATES4 2048
#define FEATN  128
#define ACTN   32
#define DN     160
#define KTOT   672     // DN + HID
#define KSTEPS 21      // 672 / 32
#define BATCH  256
#define TLEN   512

// ---- workspace layout (bytes) ----
// Wsw  : swizzled bf16 fused weights [gate4][nt32][ks21][lane32][16]  = 2,752,512 B
// h0/h1: bf16 hidden ping/pong [256*512]                              = 2x 262,144 B
// c    : f32 cell state                                               =   524,288 B
// xbf  : bf16 concat(features,actions) [B][T][160]                    = 41,943,040 B
#define WSW_BYTES  (4*32*21*32*16*2)
#define H_ELEMS    (BATCH*HID)
#define OFF_H0     (WSW_BYTES)
#define OFF_H1     (WSW_BYTES + H_ELEMS*2)
#define OFF_C      (WSW_BYTES + H_ELEMS*4)
#define OFF_X      (OFF_C + H_ELEMS*4)

// ---- gfx1250 async-to-LDS path (probe-confirmed: params are generic v4i*) ----
#if defined(__gfx1250__) && __has_builtin(__builtin_amdgcn_global_load_async_to_lds_b128)
#define HAS_ASYNC 1
#else
#define HAS_ASYNC 0
#endif

#if HAS_ASYNC
__device__ __forceinline__ void wait_async0() {
#if __has_builtin(__builtin_amdgcn_s_wait_asynccnt)
  __builtin_amdgcn_s_wait_asynccnt(0);
#else
  asm volatile("s_wait_asynccnt 0x0" ::: "memory");
#endif
}
#endif

__device__ __forceinline__ u16 f2bf(float f) {
  u32 u = __float_as_uint(f);
  u32 r = u + 0x7FFFu + ((u >> 16) & 1u);   // round-to-nearest-even
  return (u16)(r >> 16);
}
__device__ __forceinline__ float sigm(float x) {
  return 1.0f / (1.0f + __expf(-x));
}
__device__ __forceinline__ float tanh_f(float x) {
  float e = __expf(-2.0f * x);
  return (1.0f - e) / (1.0f + e);
}

// Convert W[160,2048] + U[512,2048] (f32) into bf16 blocks laid out exactly as the
// per-lane VGPR layout of the WMMA B operand (32x16, 16-bit elements).
__global__ __launch_bounds__(256) void swizzle_weights(
    const float* __restrict__ W, const float* __restrict__ U, u16* __restrict__ Wsw) {
  int idx  = blockIdx.x * 256 + threadIdx.x;       // < 1,376,256
  int j    = idx & 15;
  int lane = (idx >> 4) & 31;
  int rest = idx >> 9;                             // (g*32 + nt)*21 + s
  int s    = rest % 21;
  int blk  = rest / 21;
  int nt   = blk & 31;
  int g    = blk >> 5;
  int hh   = lane >> 4;
  int c16  = lane & 15;
  int K    = 32 * s + hh * 8 + j + (j >= 8 ? 8 : 0);
  int col  = g * HID + nt * 16 + c16;
  float v  = (K < DN) ? W[(size_t)K * GATES4 + col]
                      : U[(size_t)(K - DN) * GATES4 + col];
  Wsw[idx] = f2bf(v);
}

// One-time bf16 conversion of concat(features, actions) -> xbf[B][T][160].
__global__ __launch_bounds__(256) void prep_x(
    const float* __restrict__ feat, const float* __restrict__ act, u16* __restrict__ xbf) {
  int idx = blockIdx.x * 256 + threadIdx.x;        // < 2,621,440 (one 8-elem group each)
  int g8  = idx % 20;                              // group along D (160/8)
  int bt  = idx / 20;                              // b*512 + t
  const float* src = (g8 < 16) ? (feat + (size_t)bt * FEATN + g8 * 8)
                               : (act  + (size_t)bt * ACTN  + (size_t)(g8 - 16) * 8);
  float4 f0 = *(const float4*)src;
  float4 f1 = *(const float4*)(src + 4);
  uint4 o;
  o.x = (u32)f2bf(f0.x) | ((u32)f2bf(f0.y) << 16);
  o.y = (u32)f2bf(f0.z) | ((u32)f2bf(f0.w) << 16);
  o.z = (u32)f2bf(f1.x) | ((u32)f2bf(f1.y) << 16);
  o.w = (u32)f2bf(f1.z) | ((u32)f2bf(f1.w) << 16);
  *(uint4*)(xbf + (size_t)idx * 8) = o;            // == xbf[bt*160 + g8*8]
}

__global__ __launch_bounds__(256) void zero_state(u16* __restrict__ h0, float* __restrict__ c) {
  int idx = blockIdx.x * 256 + threadIdx.x;        // < 131072
  c[idx] = 0.0f;
  if (idx < H_ELEMS / 2) ((u32*)h0)[idx] = 0u;
}

union AVec {
  uint4    q[2];
  bf16x16  v;
};

// One LSTM timestep, fully fused: z = [x_t ; h_prev] @ [W;U] + b, gates, c/h update.
// Block = 8 waves sharing one n-column tile; B tiles flow through a double-buffered
// LDS stage fed by ASYNCcnt-tracked global_load_async_to_lds_b128 copies that
// overlap the WMMAs of the previous k-step.
__global__ __launch_bounds__(256) void lstm_step(
    const u16* __restrict__ xbf, const float* __restrict__ bias,
    const u16* __restrict__ Wsw, const u16* __restrict__ hprev,
    u16* __restrict__ hnext, float* __restrict__ cst,
    float* __restrict__ hout, int t, int last) {
  __shared__ __align__(32) uint4 ldsB[2][256];     // 2 x (4 gates x 32 lanes x 32B) = 8KB

  const int tid    = threadIdx.x;
  const int lane   = tid & 31;
  const int wave   = tid >> 5;
  const int n_tile = blockIdx.x & 31;
  const int m_tile = ((blockIdx.x >> 5) << 3) + wave;   // 0..15
  const int c16    = lane & 15;
  const int hh     = lane >> 4;
  const int arow   = m_tile * 16 + c16;                  // A-matrix row for this lane

  // Per-lane A source rows (both already bf16, contiguous along K)
  const u16* __restrict__ xrow = xbf + ((size_t)arow * TLEN + t) * DN;
  const u16* __restrict__ hrow = hprev + (size_t)arow * HID;

  // cooperative-copy coordinates for the B stage: each thread owns one 16B chunk
  const int cg = tid >> 6;         // gate 0..3
  const int cr = tid & 63;         // 16B chunk within gate block
  const u16* __restrict__ bsrc =
      Wsw + ((size_t)(cg * 32 + n_tile) * KSTEPS) * 512 + (size_t)cr * 8;

  v8f acc[4];
#pragma unroll
  for (int g = 0; g < 4; ++g)
#pragma unroll
    for (int r = 0; r < 8; ++r) acc[g][r] = 0.0f;

  // prologue: kick off k-step 0 into buffer 0
#if HAS_ASYNC
  __builtin_amdgcn_global_load_async_to_lds_b128(
      (v4i*)bsrc, (v4i*)&ldsB[0][tid], 0, 0);
#else
  ldsB[0][tid] = *(const uint4*)bsrc;
#endif

  for (int s = 0; s < KSTEPS; ++s) {
    const int cur = s & 1;
#if HAS_ASYNC
    wait_async0();               // my chunk of buf[cur] has landed
#endif
    __syncthreads();             // all chunks landed; all waves done reading buf[!cur]

    if (s + 1 < KSTEPS) {        // overlap next B copy with this step's WMMAs
#if HAS_ASYNC
      __builtin_amdgcn_global_load_async_to_lds_b128(
          (v4i*)(bsrc + (size_t)(s + 1) * 512), (v4i*)&ldsB[cur ^ 1][tid], 0, 0);
#else
      ldsB[cur ^ 1][tid] = *(const uint4*)(bsrc + (size_t)(s + 1) * 512);
#endif
    }
    if (s + 2 < KSTEPS)
      __builtin_prefetch(bsrc + (size_t)(s + 2) * 512, 0, 1);

    // assemble A operand (16x32 bf16 tile, per-lane layout) -- two b128 loads
    int g1 = 4 * s + hh;
    int g2 = g1 + 2;
    AVec a;
    a.q[0] = *(const uint4*)((g1 < 20) ? (xrow + g1 * 8) : (hrow + (g1 - 20) * 8));
    a.q[1] = *(const uint4*)((g2 < 20) ? (xrow + g2 * 8) : (hrow + (g2 - 20) * 8));

    const bf16x16* lb = (const bf16x16*)ldsB[cur];
#pragma unroll
    for (int g = 0; g < 4; ++g) {
      bf16x16 bm = lb[g * 32 + lane];
      acc[g] = __builtin_amdgcn_wmma_f32_16x16x32_bf16(
          false, a.v, false, bm, (short)0, acc[g], false, false);
    }
  }

  // gate math + state update (each (b,n) owned by exactly one wave)
  const int n = n_tile * 16 + c16;
  const float bi = bias[n];
  const float bf = bias[HID + n];
  const float bg = bias[2 * HID + n];
  const float bo = bias[3 * HID + n];

#pragma unroll
  for (int r = 0; r < 8; ++r) {
    int brow = m_tile * 16 + hh * 8 + r;
    size_t idx = (size_t)brow * HID + n;
    float zi = acc[0][r] + bi;
    float zf = acc[1][r] + bf;
    float zg = acc[2][r] + bg;
    float zo = acc[3][r] + bo;
    float iS = sigm(zi), fS = sigm(zf), gT = tanh_f(zg), oS = sigm(zo);
    float cn = fS * cst[idx] + iS * gT;
    float hn = oS * tanh_f(cn);
    cst[idx]   = cn;
    hnext[idx] = f2bf(hn);
    if (last) hout[idx] = hn;
  }
}

extern "C" void kernel_launch(void* const* d_in, const int* in_sizes, int n_in,
                              void* d_out, int out_size, void* d_ws, size_t ws_size,
                              hipStream_t stream) {
  const float* feat = (const float*)d_in[0];
  const float* act  = (const float*)d_in[1];
  const float* W    = (const float*)d_in[2];
  const float* U    = (const float*)d_in[3];
  const float* bias = (const float*)d_in[4];
  float* hout = (float*)d_out;

  char* ws = (char*)d_ws;
  u16*   Wsw = (u16*)ws;
  u16*   h0  = (u16*)(ws + OFF_H0);
  u16*   h1  = (u16*)(ws + OFF_H1);
  float* cst = (float*)(ws + OFF_C);
  u16*   xbf = (u16*)(ws + OFF_X);

  swizzle_weights<<<5376, 256, 0, stream>>>(W, U, Wsw);   // 1,376,256 elems
  prep_x<<<10240, 256, 0, stream>>>(feat, act, xbf);      // 2,621,440 groups
  zero_state<<<512, 256, 0, stream>>>(h0, cst);

  for (int t = 0; t < TLEN; ++t) {
    u16* hp = (t & 1) ? h1 : h0;
    u16* hn = (t & 1) ? h0 : h1;
    lstm_step<<<64, 256, 0, stream>>>(xbf, bias, Wsw, hp, hn, cst, hout,
                                      t, t == TLEN - 1 ? 1 : 0);
  }
}
